// Relative_Attention_13451837571903
// MI455X (gfx1250) — compile-verified, matrix operand order checked by reading, not run on hardware
//
#include <hip/hip_runtime.h>
#include <hip/hip_bf16.h>
#include <math.h>

#define BB 16
#define NN 577
#define CC 768
#define HH 12
#define HD 64
#define NPAD 608            // 19 * 32 key blocks
#define MROWS (BB*NN)       // 9232
#define MTILES (MROWS/16)   // 577
#define ITILES 37           // ceil(577/16)

#if __has_builtin(__builtin_amdgcn_global_load_async_to_lds_b128) && \
    __has_builtin(__builtin_amdgcn_s_wait_asynccnt)
#define HAVE_ASYNC_LDS 1
#else
#define HAVE_ASYNC_LDS 0
#endif

typedef __attribute__((ext_vector_type(16))) __bf16 v16bf;
typedef __attribute__((ext_vector_type(8)))  float  v8f;
typedef __attribute__((ext_vector_type(4)))  int    v4i;
typedef unsigned short u16;
typedef unsigned int   u32;

#if HAVE_ASYNC_LDS
typedef __attribute__((address_space(1))) v4i  gv4i;
typedef __attribute__((address_space(3))) v4i  lv4i;
typedef __attribute__((address_space(1))) void gvoid;
typedef __attribute__((address_space(3))) void lvoid;
#endif

union BF16Frag { v16bf v; u16 u[16]; u32 w[8]; };

__device__ __forceinline__ u16 f2bf(float f){
  u32 u = __builtin_bit_cast(u32, f);
  u32 r = u + 0x7FFFu + ((u >> 16) & 1u);
  return (u16)(r >> 16);
}
__device__ __forceinline__ float bf2f(u16 h){
  u32 u = ((u32)h) << 16;
  return __builtin_bit_cast(float, u);
}
// K offset pattern for 16-bit 16x32 A/B fragments (ISA 7.12.2), hf = lane>>4
__device__ __forceinline__ int kboff(int i, int hf){
  return (i < 4) ? (hf*8 + 2*i) : (16 + hf*8 + 2*(i-4));
}
// Lane reads one row of a row-major bf16 matrix (A rows, or B^T rows = W rows)
__device__ __forceinline__ v16bf load_row_frag(const u16* __restrict__ rowp, int k0, int hf){
  BF16Frag f;
#pragma unroll
  for (int i = 0; i < 8; ++i){
    int kb = k0 + kboff(i, hf);
    f.w[i] = *(const u32*)(rowp + kb);           // two adjacent bf16
  }
  return f.v;
}
// B fragment for P@V from an LDS 32x64 V tile: lane owns column dcol
__device__ __forceinline__ v16bf load_v_frag(const u16* vt, int dcol, int hf){
  BF16Frag f;
#pragma unroll
  for (int i = 0; i < 8; ++i){
    int kb = kboff(i, hf);
    f.u[2*i]   = vt[kb*HD + dcol];
    f.u[2*i+1] = vt[(kb+1)*HD + dcol];
  }
  return f.v;
}
// A fragment (16x32) from an LDS row-major 16x32 P tile
__device__ __forceinline__ v16bf load_p_frag(const u16* pt, int m, int hf){
  BF16Frag f;
#pragma unroll
  for (int i = 0; i < 8; ++i)
    f.w[i] = *(const u32*)(pt + m*32 + kboff(i, hf));
  return f.v;
}
__device__ __forceinline__ v8f wmma_bf16(v16bf a, v16bf b, v8f c){
  return __builtin_amdgcn_wmma_f32_16x16x32_bf16(false, a, false, b, (short)0, c, false, false);
}

// ---------------------------------------------------------------- cast
__global__ void cast_f32_bf16(const float* __restrict__ src, u16* __restrict__ dst, int n){
  int i = blockIdx.x*blockDim.x + threadIdx.x;
  int stride = gridDim.x*blockDim.x;
  for (; i < n; i += stride) dst[i] = f2bf(src[i]);
}

// zero padded rows 577..607 of q/k/v
__global__ void zero_pad_rows(u16* q, u16* k, u16* v){
  int bh = blockIdx.x;
  u16* buf = (blockIdx.y == 0) ? q : ((blockIdx.y == 1) ? k : v);
  u16* base = buf + ((size_t)bh*NPAD + NN)*HD;
  for (int i = threadIdx.x; i < (NPAD-NN)*HD; i += blockDim.x) base[i] = 0;
}

// ---------------------------------------------------------------- QKV GEMM
// out = x_bf16 @ qkv_w^T ; scatter into q/k/v (B*H, NPAD, 64), q scaled by 1/8
__global__ void __launch_bounds__(128) gemm_qkv(const u16* __restrict__ xb, const u16* __restrict__ wb,
                                                u16* __restrict__ q, u16* __restrict__ k, u16* __restrict__ v){
  int lane = threadIdx.x & 31;
  int wave = threadIdx.x >> 5;
  int hf = lane >> 4;
  int nl = lane & 15;
  int row_base = blockIdx.x * 16;
  int col_base = (blockIdx.y * 4 + wave) * 16;
  const u16* arow = xb + (size_t)(row_base + nl)*CC;
  const u16* brow = wb + (size_t)(col_base + nl)*CC;
  v8f acc = {};
  for (int k0 = 0; k0 < CC; k0 += 32){
    __builtin_prefetch(arow + k0 + 64, 0, 3);   // WGP-scope temporal prefetch
    __builtin_prefetch(brow + k0 + 64, 0, 3);
    acc = wmma_bf16(load_row_frag(arow, k0, hf), load_row_frag(brow, k0, hf), acc);
  }
  int gc = col_base + nl;           // column in [0, 2304)
  int which = gc / CC;
  int c = gc % CC;
  int h = c / HD, d = c % HD;
  float scale = (which == 0) ? 0.125f : 1.0f;  // hd^-0.5 = 1/8
  u16* dst = (which == 0) ? q : ((which == 1) ? k : v);
#pragma unroll
  for (int j = 0; j < 8; ++j){
    int mg = row_base + j + 8*hf;
    int b_ = mg / NN, tok = mg % NN;
    dst[((size_t)(b_*HH + h)*NPAD + tok)*HD + d] = f2bf(acc[j]*scale);
  }
}

// ---------------------------------------------------------------- flash attention
// 1 wave per 16-row q tile; 8 waves/block share (b,h); 32-key blocks.
__global__ void __launch_bounds__(256) flash_attn(
    const u16* __restrict__ qb, const u16* __restrict__ kb, const u16* __restrict__ vb,
    const float* __restrict__ bias_table, const int* __restrict__ rel_index,
    u16* __restrict__ attn_out)
{
  __shared__ __align__(16) u16 Vt[32*HD];       // 4 KB shared V block
  __shared__ __align__(16) u16 Pt[8][16*32];    // 1 KB per wave P tile
  int tid  = threadIdx.x;
  int lane = tid & 31, wave = tid >> 5;
  int hf = lane >> 4, nl = lane & 15;
  int bh = blockIdx.x;
  int h = bh % HH;
  int i_tile = blockIdx.y * 8 + wave;
  bool wvalid = (i_tile < ITILES);
  int ibase = i_tile * 16;
  const u16* qmat = qb + (size_t)bh * NPAD * HD;
  const u16* kmat = kb + (size_t)bh * NPAD * HD;
  const u16* vmat = vb + (size_t)bh * NPAD * HD;
  // q fragments are j-loop invariant (16x64)
  const u16* qrow = qmat + (size_t)((wvalid ? ibase : 0) + nl) * HD;
  v16bf qa0 = load_row_frag(qrow, 0, hf);
  v16bf qa1 = load_row_frag(qrow, 32, hf);
  v8f o0 = {}, o1 = {}, o2 = {}, o3 = {};
  float mrun[8], lrun[8];
#pragma unroll
  for (int j = 0; j < 8; ++j){ mrun[j] = -1e30f; lrun[j] = 0.0f; }

  for (int jb = 0; jb < NPAD; jb += 32){
    // stage V block (32x64 bf16 = 256 x 16B) into LDS
#if HAVE_ASYNC_LDS
    // CDNA5 async copy: global -> LDS without touching VGPRs (ASYNCcnt tracked)
    __builtin_amdgcn_global_load_async_to_lds_b128(
        (gv4i*)(gvoid*)(const void*)(vmat + (size_t)jb*HD + tid*8),
        (lv4i*)(lvoid*)(void*)(Vt + tid*8),
        0, 0);
    __builtin_amdgcn_s_wait_asynccnt(0);
#else
    ((uint4*)Vt)[tid] = ((const uint4*)(vmat + (size_t)jb*HD))[tid];
#endif
    __syncthreads();
    if (wvalid){
      const u16* krow0 = kmat + (size_t)(jb + nl) * HD;
      const u16* krow1 = krow0 + 16*HD;
      v8f c0 = {}, c1 = {};
      c0 = wmma_bf16(qa0, load_row_frag(krow0, 0,  hf), c0);
      c0 = wmma_bf16(qa1, load_row_frag(krow0, 32, hf), c0);
      c1 = wmma_bf16(qa0, load_row_frag(krow1, 0,  hf), c1);
      c1 = wmma_bf16(qa1, load_row_frag(krow1, 32, hf), c1);
      int n0 = jb + nl, n1 = n0 + 16;
      u16* pt = Pt[wave];
#pragma unroll
      for (int j = 0; j < 8; ++j){
        int m = j + 8*hf;
        int qi = ibase + m;
        float s0 = -1e30f, s1 = -1e30f;
        if (qi < NN){
          if (n0 < NN) s0 = c0[j] + bias_table[rel_index[qi*NN + n0]*HH + h];
          if (n1 < NN) s1 = c1[j] + bias_table[rel_index[qi*NN + n1]*HH + h];
        }
        float rmax = fmaxf(s0, s1);                 // row reduce across 16 lanes
        rmax = fmaxf(rmax, __shfl_xor(rmax, 1, 32));
        rmax = fmaxf(rmax, __shfl_xor(rmax, 2, 32));
        rmax = fmaxf(rmax, __shfl_xor(rmax, 4, 32));
        rmax = fmaxf(rmax, __shfl_xor(rmax, 8, 32));
        float mnew = fmaxf(mrun[j], rmax);
        float fsc  = __expf(mrun[j] - mnew);
        float p0 = __expf(s0 - mnew);
        float p1 = __expf(s1 - mnew);
        float rs = p0 + p1;
        rs += __shfl_xor(rs, 1, 32);
        rs += __shfl_xor(rs, 2, 32);
        rs += __shfl_xor(rs, 4, 32);
        rs += __shfl_xor(rs, 8, 32);
        lrun[j] = lrun[j]*fsc + rs;
        mrun[j] = mnew;
        o0[j] *= fsc; o1[j] *= fsc; o2[j] *= fsc; o3[j] *= fsc;
        pt[m*32 + nl]      = f2bf(p0);
        pt[m*32 + 16 + nl] = f2bf(p1);
      }
      asm volatile("s_wait_dscnt 0" ::: "memory");  // wave-local LDS RAW fence
      v16bf pa = load_p_frag(pt, nl, hf);           // P as A-fragment (16x32)
      o0 = wmma_bf16(pa, load_v_frag(Vt,  0 + nl, hf), o0);
      o1 = wmma_bf16(pa, load_v_frag(Vt, 16 + nl, hf), o1);
      o2 = wmma_bf16(pa, load_v_frag(Vt, 32 + nl, hf), o2);
      o3 = wmma_bf16(pa, load_v_frag(Vt, 48 + nl, hf), o3);
    }
    __syncthreads();
  }
  if (wvalid){
    int b_ = bh / HH;
#pragma unroll
    for (int j = 0; j < 8; ++j){
      int qi = ibase + j + 8*hf;
      if (qi >= NN) continue;
      float inv = (lrun[j] > 0.0f) ? 1.0f/lrun[j] : 0.0f;
      size_t rowa = ((size_t)b_*NN + qi)*CC + h*HD;
      attn_out[rowa +  0 + nl] = f2bf(o0[j]*inv);
      attn_out[rowa + 16 + nl] = f2bf(o1[j]*inv);
      attn_out[rowa + 32 + nl] = f2bf(o2[j]*inv);
      attn_out[rowa + 48 + nl] = f2bf(o3[j]*inv);
    }
  }
}

// ---------------------------------------------------------------- CLS row attention (for selection)
__global__ void __launch_bounds__(256) cls_row0(
    const u16* __restrict__ qb, const u16* __restrict__ kbuf,
    const float* __restrict__ bias_table, const int* __restrict__ rel_index,
    float* __restrict__ cls_probs)
{
  __shared__ float sbuf[NN];
  __shared__ float red[256];
  int bh = blockIdx.x, h = bh % HH;
  const u16* qrow = qb + (size_t)bh*NPAD*HD;
  const u16* kmat = kbuf + (size_t)bh*NPAD*HD;
  int tid = threadIdx.x;
  float lmax = -1e30f;
  for (int j = tid; j < NN; j += 256){
    float s = 0.0f;
    const u16* krow = kmat + (size_t)j*HD;
    for (int d = 0; d < HD; ++d) s += bf2f(qrow[d])*bf2f(krow[d]);
    s += bias_table[rel_index[j]*HH + h];
    sbuf[j] = s;
    lmax = fmaxf(lmax, s);
  }
  red[tid] = lmax; __syncthreads();
  for (int st = 128; st > 0; st >>= 1){ if (tid < st) red[tid] = fmaxf(red[tid], red[tid+st]); __syncthreads(); }
  float mx = red[0]; __syncthreads();
  float lsum = 0.0f;
  for (int j = tid; j < NN; j += 256){ float e = __expf(sbuf[j]-mx); sbuf[j] = e; lsum += e; }
  red[tid] = lsum; __syncthreads();
  for (int st = 128; st > 0; st >>= 1){ if (tid < st) red[tid] += red[tid+st]; __syncthreads(); }
  float inv = 1.0f/red[0];
  for (int j = tid; j < NN; j += 256) cls_probs[(size_t)bh*NN + j] = sbuf[j]*inv;
}

__global__ void cls_stats(const float* __restrict__ cls_probs,
                          float* __restrict__ cls_attn, int* __restrict__ counts)
{
  __shared__ float sb[576];
  int b = blockIdx.x, j = threadIdx.x;           // blockDim = 576
  float v = 0.0f;
  for (int h = 0; h < HH; ++h) v += cls_probs[(size_t)(b*HH+h)*NN + 1 + j];
  v *= (1.0f/HH);
  cls_attn[b*576 + j] = v;
  sb[j] = v;
  __syncthreads();
  if (j == 0){
    float s = 0.0f, mx = -1e30f;
    for (int t = 0; t < 576; ++t){ s += sb[t]; mx = fmaxf(mx, sb[t]); }
    float avg = s/576.0f;
    int cnt = 0;
    for (int t = 0; t < 576; ++t) if (sb[t] >= avg && sb[t] <= mx) ++cnt;
    counts[b] = cnt;
  }
}

__global__ void keep_info(const int* __restrict__ counts, const float* __restrict__ tw,
                          int* __restrict__ ki)
{
  if (threadIdx.x != 0) return;
  int mc = 0; float sum = 0.0f;
  for (int b = 0; b < BB; ++b){ sum += (float)counts[b]; mc = counts[b] > mc ? counts[b] : mc; }
  float T = sum / (float)BB;
  float sig = 1.0f/(1.0f + __expf(-(((float)mc) - T)/(float)BB));
  int keep = (int)rintf(576.0f*(0.6f + 0.4f*sig));
  if (keep < 0) keep = 0; if (keep > 576) keep = 576;
  int remaining = NN - 1 - keep;
  float e0 = __expf(tw[0]), e1 = __expf(tw[1]);
  float w0 = e0/(e0+e1);
  int num_text = (int)(w0 * (float)remaining);
  ki[0] = keep; ki[1] = num_text; ki[2] = remaining - num_text;
}

// stable descending rank == jax.lax.top_k order
__global__ void rank_topk(const float* __restrict__ cls_attn, int* __restrict__ order)
{
  __shared__ float sb[576];
  int b = blockIdx.x, j = threadIdx.x;
  float v = cls_attn[b*576 + j];
  sb[j] = v; __syncthreads();
  int rank = 0;
  for (int t = 0; t < 576; ++t){
    float u = sb[t];
    rank += (u > v) || (u == v && t < j);
  }
  order[b*576 + rank] = j;
}

__global__ void dedupe(const int* __restrict__ order, const int* __restrict__ ki,
                       int* __restrict__ keep_idx)
{
  if (threadIdx.x != 0) return;
  bool seen[576];
  for (int t = 0; t < 576; ++t) seen[t] = false;
  int keep = ki[0], cnt = 0;
  for (int b = 0; b < BB && cnt < keep; ++b)
    for (int r = 0; r < keep && cnt < keep; ++r){
      int j = order[b*576 + r];
      if (!seen[j]){ seen[j] = true; keep_idx[cnt++] = j; }
    }
}

// ---------------------------------------------------------------- proj GEMM
__global__ void __launch_bounds__(128) gemm_proj(const u16* __restrict__ ao, const u16* __restrict__ wpb,
                                                 const float* __restrict__ bias, float* __restrict__ xo)
{
  int lane = threadIdx.x & 31, wave = threadIdx.x >> 5;
  int hf = lane >> 4, nl = lane & 15;
  int row_base = blockIdx.x*16;
  int col_base = (blockIdx.y*4 + wave)*16;
  const u16* arow = ao  + (size_t)(row_base + nl)*CC;
  const u16* brow = wpb + (size_t)(col_base + nl)*CC;
  v8f acc = {};
  for (int k0 = 0; k0 < CC; k0 += 32){
    __builtin_prefetch(arow + k0 + 64, 0, 3);
    acc = wmma_bf16(load_row_frag(arow, k0, hf), load_row_frag(brow, k0, hf), acc);
  }
  int gc = col_base + nl;
  float bv = bias[gc];
#pragma unroll
  for (int j = 0; j < 8; ++j){
    int mg = row_base + j + 8*hf;
    xo[(size_t)mg*CC + gc] = acc[j] + bv;
  }
}

// ---------------------------------------------------------------- assemble
__global__ void assemble(const float* __restrict__ xo, const float* __restrict__ xin,
                         const int* __restrict__ ki, const int* __restrict__ keep_idx,
                         float* __restrict__ out)
{
  int t = blockIdx.x, b = blockIdx.y;
  int keep = ki[0], num_text = ki[1];
  const float* src;
  if (t == 0)                         src = xo  + (size_t)(b*NN)*CC;
  else if (t <= keep)                 src = xo  + (size_t)(b*NN + 1 + keep_idx[t-1])*CC;
  else if (t <= keep + num_text)      src = xin + (size_t)(b*NN + 1)*CC;     // EXTRA=1 -> idx 0
  else {
    int ti = t - 1 - keep - num_text;
    unsigned r = (unsigned)(b*1315423911u) ^ (unsigned)(ti*2654435761u);
    r ^= r >> 13; r *= 0x5bd1e995u; r ^= r >> 15;
    src = xin + (size_t)(b*NN + 2 + (int)(r % 575u))*CC;
  }
  float* dst = out + (size_t)(b*NN + t)*CC;
  for (int c = threadIdx.x; c < CC; c += blockDim.x) dst[c] = src[c];
}

// ---------------------------------------------------------------- launch
extern "C" void kernel_launch(void* const* d_in, const int* in_sizes, int n_in,
                              void* d_out, int out_size, void* d_ws, size_t ws_size,
                              hipStream_t stream)
{
  (void)in_sizes; (void)n_in; (void)out_size; (void)ws_size;
  const float* x     = (const float*)d_in[0];
  const float* qkvw  = (const float*)d_in[1];
  const float* projw = (const float*)d_in[2];
  const float* projb = (const float*)d_in[3];
  const float* btab  = (const float*)d_in[4];
  const float* tw    = (const float*)d_in[5];
  const int*   ridx  = (const int*)d_in[6];
  float* out = (float*)d_out;

  char* ws = (char*)d_ws;
  size_t off = 0;
  auto alloc = [&](size_t bytes)->char*{
    char* p = ws + off; off += (bytes + 255) & ~(size_t)255; return p;
  };
  u16*   xb    = (u16*)  alloc((size_t)MROWS*CC*2);
  u16*   wqb   = (u16*)  alloc((size_t)3*CC*CC*2);
  u16*   wpb   = (u16*)  alloc((size_t)CC*CC*2);
  u16*   qbuf  = (u16*)  alloc((size_t)BB*HH*NPAD*HD*2);
  u16*   kbuf  = (u16*)  alloc((size_t)BB*HH*NPAD*HD*2);
  u16*   vbuf  = (u16*)  alloc((size_t)BB*HH*NPAD*HD*2);
  u16*   aob   = (u16*)  alloc((size_t)MROWS*CC*2);
  float* xo    = (float*)alloc((size_t)MROWS*CC*4);
  float* clsp  = (float*)alloc((size_t)BB*HH*NN*4);
  float* clsa  = (float*)alloc((size_t)BB*576*4);
  int*   counts= (int*)  alloc(BB*4);
  int*   ki    = (int*)  alloc(64);
  int*   order = (int*)  alloc((size_t)BB*576*4);
  int*   kidx  = (int*)  alloc(576*4);

  cast_f32_bf16<<<512, 256, 0, stream>>>(x,     xb,  MROWS*CC);
  cast_f32_bf16<<<256, 256, 0, stream>>>(qkvw,  wqb, 3*CC*CC);
  cast_f32_bf16<<<128, 256, 0, stream>>>(projw, wpb, CC*CC);
  zero_pad_rows<<<dim3(BB*HH, 3), 256, 0, stream>>>(qbuf, kbuf, vbuf);
  gemm_qkv<<<dim3(MTILES, 36), 128, 0, stream>>>(xb, wqb, qbuf, kbuf, vbuf);
  flash_attn<<<dim3(BB*HH, 5), 256, 0, stream>>>(qbuf, kbuf, vbuf, btab, ridx, aob);
  cls_row0<<<BB*HH, 256, 0, stream>>>(qbuf, kbuf, btab, ridx, clsp);
  cls_stats<<<BB, 576, 0, stream>>>(clsp, clsa, counts);
  keep_info<<<1, 32, 0, stream>>>(counts, tw, ki);
  rank_topk<<<BB, 576, 0, stream>>>(clsa, order);
  dedupe<<<1, 32, 0, stream>>>(order, ki, kidx);
  gemm_proj<<<dim3(MTILES, 12), 128, 0, stream>>>(aob, wpb, projb, xo);
  assemble<<<dim3(NN, BB), 256, 0, stream>>>(xo, x, ki, kidx, out);
}
